// MultiHeadAttention_8684423872737
// MI455X (gfx1250) — matrix operand order repaired and checked by reading, not compiled, not run on hardware
//
#include <hip/hip_runtime.h>

#define DIM   2048
#define TSEQ  2048
#define NHQ   32
#define NHKV  8
#define HEADD 64
#define KVDIM (NHKV * HEADD)  // 512

typedef __bf16 bf16_t;
typedef __attribute__((ext_vector_type(16))) __bf16 v16bf;
typedef __attribute__((ext_vector_type(8)))  __bf16 v8bf;
typedef __attribute__((ext_vector_type(8)))  float  v8f;

#define WMMA_BF16(Afrag, Bfrag, Cacc) \
  __builtin_amdgcn_wmma_f32_16x16x32_bf16(false, (Afrag), false, (Bfrag), (short)0, (Cacc), false, false)

// Build a 16-element bf16 WMMA fragment from two contiguous 16-byte chunks.
static __device__ __forceinline__ v16bf frag16(const bf16_t* lo, const bf16_t* hi) {
  v8bf a = *(const v8bf*)lo;
  v8bf b = *(const v8bf*)hi;
  v16bf r;
#pragma unroll
  for (int e = 0; e < 8; ++e) { r[e] = a[e]; r[e + 8] = b[e]; }
  return r;
}

// ---------------- conversion kernels ----------------
__global__ __launch_bounds__(256) void cvt_f32_bf16(const float* __restrict__ in,
                                                    bf16_t* __restrict__ out, int n) {
  int i = blockIdx.x * 256 + threadIdx.x;
  if (i < n) out[i] = (bf16_t)in[i];
}

// out[n*K + k] = in[k*N + n]  (store B^T so WMMA B-fragments are contiguous)
__global__ __launch_bounds__(256) void cvtT_f32_bf16(const float* __restrict__ in,
                                                     bf16_t* __restrict__ out, int K, int N) {
  int i = blockIdx.x * 256 + threadIdx.x;
  if (i < K * N) {
    int k = i / N, n = i % N;
    out[(size_t)n * K + k] = (bf16_t)in[i];
  }
}

// ---------------- GEMM: C = A(bf16,[M,K]) * Bt^T + bias ----------------
// Each wave: 16x64 tile, register double-buffered fragments so next K-step's
// VMEM overlaps the current 4 WMMAs.
// mode 0: fp32 row-major (out proj)   mode 1: bf16 row-major (Q)
// mode 2: K: fp32+bf16 [Hkv,T,64]     mode 3: V: fp32 [Hkv,T,64], bf16 [Hkv*64,T]
__global__ __launch_bounds__(256) void gemm_bf16_wmma(
    const bf16_t* __restrict__ A, const bf16_t* __restrict__ Bt,
    const float* __restrict__ bias, float* __restrict__ Cf, bf16_t* __restrict__ Cb,
    int M, int N, int K, int mode) {
  const int lane = threadIdx.x & 31;
  const int half = lane >> 4;
  const int ln   = lane & 15;
  const int row0 = (blockIdx.y * 8 + (threadIdx.x >> 5)) * 16;
  const int col0 = blockIdx.x * 64;
  if (row0 >= M) return;  // wave-uniform

  const bf16_t* ar  = A  + (size_t)(row0 + ln) * K;
  const bf16_t* br0 = Bt + (size_t)(col0 + ln) * K;
  const bf16_t* br1 = br0 + (size_t)16 * K;
  const bf16_t* br2 = br0 + (size_t)32 * K;
  const bf16_t* br3 = br0 + (size_t)48 * K;
  const int ao = half * 16;  // A-frag: elems 0..15 at k+ao
  const int bo = half * 16;  // B-frag: elems 0..15 at k+bo

  v8f acc[4] = {};
  v16bf ca  = frag16(ar  + ao, ar  + ao + 8);
  v16bf cb0 = frag16(br0 + bo, br0 + bo + 8);
  v16bf cb1 = frag16(br1 + bo, br1 + bo + 8);
  v16bf cb2 = frag16(br2 + bo, br2 + bo + 8);
  v16bf cb3 = frag16(br3 + bo, br3 + bo + 8);

  for (int k0 = 0; k0 < K - 32; k0 += 32) {
    const int kn = k0 + 32;
    // issue next step's loads first, then consume current frags
    v16bf na  = frag16(ar  + kn + ao, ar  + kn + ao + 8);
    v16bf nb0 = frag16(br0 + kn + bo, br0 + kn + bo + 8);
    v16bf nb1 = frag16(br1 + kn + bo, br1 + kn + bo + 8);
    v16bf nb2 = frag16(br2 + kn + bo, br2 + kn + bo + 8);
    v16bf nb3 = frag16(br3 + kn + bo, br3 + kn + bo + 8);
    __builtin_prefetch(ar + kn + 256, 0, 1);
    acc[0] = WMMA_BF16(ca, cb0, acc[0]);
    acc[1] = WMMA_BF16(ca, cb1, acc[1]);
    acc[2] = WMMA_BF16(ca, cb2, acc[2]);
    acc[3] = WMMA_BF16(ca, cb3, acc[3]);
    ca = na; cb0 = nb0; cb1 = nb1; cb2 = nb2; cb3 = nb3;
  }
  acc[0] = WMMA_BF16(ca, cb0, acc[0]);
  acc[1] = WMMA_BF16(ca, cb1, acc[1]);
  acc[2] = WMMA_BF16(ca, cb2, acc[2]);
  acc[3] = WMMA_BF16(ca, cb3, acc[3]);

#pragma unroll
  for (int t = 0; t < 4; ++t) {
    int c = col0 + t * 16 + ln;
    float bv = bias ? bias[c] : 0.0f;
#pragma unroll
    for (int i = 0; i < 8; ++i) {
      int r = row0 + half * 8 + i;  // C layout: VGPR i -> row i / 8+i by half
      float v = acc[t][i] + bv;
      if (mode == 0) {
        Cf[(size_t)r * N + c] = v;
      } else if (mode == 1) {
        Cb[(size_t)r * N + c] = (bf16_t)v;
      } else if (mode == 2) {
        int hh = c >> 6, dd = c & 63;
        size_t off = ((size_t)hh * TSEQ + r) * HEADD + dd;
        Cf[off] = v;
        Cb[off] = (bf16_t)v;
      } else {
        int hh = c >> 6, dd = c & 63;
        Cf[((size_t)hh * TSEQ + r) * HEADD + dd] = v;
        Cb[(size_t)(hh * HEADD + dd) * TSEQ + r] = (bf16_t)v;
      }
    }
  }
}

// ---------------- causal GQA flash attention ----------------
// One wave per (head, 16-row query tile). Score WMMAs run on preloaded K frags;
// V frags + next K frags are issued before the softmax VALU chain so VMEM
// latency hides under it. Probs staged via LDS (C layout -> A layout).
__global__ __launch_bounds__(32) void flash_attn_gqa(
    const bf16_t* __restrict__ Qb,  // [T, DIM] bf16
    const bf16_t* __restrict__ Kb,  // [NHKV, T, HEADD] bf16
    const bf16_t* __restrict__ Vt,  // [NHKV*HEADD, T] bf16 (V transposed)
    bf16_t* __restrict__ Ob) {      // [T, DIM] bf16
  __shared__ bf16_t smem[16 * 32];
  const int lane = threadIdx.x & 31;
  const int half = lane >> 4;
  const int ln   = lane & 15;
  const int h    = blockIdx.x & (NHQ - 1);
  const int row0 = (blockIdx.x >> 5) * 16;
  const int kvh  = h >> 2;  // GQA group = 4

  const bf16_t* qrow = Qb + (size_t)(row0 + ln) * DIM + h * HEADD;
  v16bf qa0 = frag16(qrow + half * 16,      qrow + half * 16 + 8);       // d 0..31
  v16bf qa1 = frag16(qrow + 32 + half * 16, qrow + 32 + half * 16 + 8);  // d 32..63

  float m[8], l[8];
  v8f acc[4] = {};
#pragma unroll
  for (int i = 0; i < 8; ++i) { m[i] = -1e30f; l[i] = 0.0f; }

  const int qend   = row0 + 15;
  const int kblast = (qend >> 5) << 5;  // last 32-key block start

  const bf16_t* kbase = Kb + (size_t)kvh * TSEQ * HEADD;
  // preload K frags for block kb = 0 (two 16-key tiles x two 32-d chunks)
  v16bf k0lo, k0hi, k1lo, k1hi;
  {
    const bf16_t* kp0 = kbase + (size_t)ln * HEADD + half * 16;
    const bf16_t* kp1 = kp0 + 16 * HEADD;
    k0lo = frag16(kp0, kp0 + 8);  k0hi = frag16(kp0 + 32, kp0 + 40);
    k1lo = frag16(kp1, kp1 + 8);  k1hi = frag16(kp1 + 32, kp1 + 40);
  }

  for (int kb = 0;; kb += 32) {
    const bool last = (kb == kblast);  // wave-uniform
    // scores: S[16q x 32keys] via 4 WMMAs on preloaded frags
    v8f s0 = {}, s1 = {};
    s0 = WMMA_BF16(qa0, k0lo, s0);
    s0 = WMMA_BF16(qa1, k0hi, s0);
    s1 = WMMA_BF16(qa0, k1lo, s1);
    s1 = WMMA_BF16(qa1, k1hi, s1);

    // issue this block's V frags + next block's K frags; they complete under softmax
    v16bf vf[4];
#pragma unroll
    for (int t = 0; t < 4; ++t) {
      const bf16_t* vp = Vt + (size_t)(kvh * HEADD + t * 16 + ln) * TSEQ + kb + half * 16;
      vf[t] = frag16(vp, vp + 8);
    }
    v16bf n0lo = {}, n0hi = {}, n1lo = {}, n1hi = {};
    if (!last) {
      const bf16_t* kp0 = kbase + (size_t)(kb + 32 + ln) * HEADD + half * 16;
      const bf16_t* kp1 = kp0 + 16 * HEADD;
      n0lo = frag16(kp0, kp0 + 8);  n0hi = frag16(kp0 + 32, kp0 + 40);
      n1lo = frag16(kp1, kp1 + 8);  n1hi = frag16(kp1 + 32, kp1 + 40);
    }

    // online softmax (rows live across 16-lane halves; butterfly reductions)
#pragma unroll
    for (int i = 0; i < 8; ++i) {
      const int qr = row0 + half * 8 + i;
      float v0 = s0[i] * 0.125f;  // 1/sqrt(64)
      float v1 = s1[i] * 0.125f;
      v0 = ((kb + ln) > qr) ? -1e30f : v0;
      v1 = ((kb + 16 + ln) > qr) ? -1e30f : v1;
      float loc = fmaxf(v0, v1);
#pragma unroll
      for (int off = 8; off >= 1; off >>= 1) loc = fmaxf(loc, __shfl_xor(loc, off, 32));
      const float mn = fmaxf(m[i], loc);
      const float p0 = __expf(v0 - mn);
      const float p1 = __expf(v1 - mn);
      float rs = p0 + p1;
#pragma unroll
      for (int off = 8; off >= 1; off >>= 1) rs += __shfl_xor(rs, off, 32);
      const float corr = __expf(m[i] - mn);
      l[i] = l[i] * corr + rs;
      m[i] = mn;
#pragma unroll
      for (int t = 0; t < 4; ++t) acc[t][i] *= corr;
      smem[(half * 8 + i) * 32 + ln]      = (bf16_t)p0;  // [qrow][key-in-block]
      smem[(half * 8 + i) * 32 + 16 + ln] = (bf16_t)p1;
    }
    asm volatile("s_wait_dscnt 0x0" ::: "memory");  // cross-lane LDS RAW within wave
    const bf16_t* pp = smem + ln * 32 + half * 16;  // A-frag layout of probs
    v16bf pf = frag16(pp, pp + 8);
#pragma unroll
    for (int t = 0; t < 4; ++t) acc[t] = WMMA_BF16(pf, vf[t], acc[t]);
    asm volatile("" ::: "memory");  // keep next-iter smem stores after these reads

    if (last) break;
    k0lo = n0lo; k0hi = n0hi; k1lo = n1lo; k1hi = n1hi;
  }

#pragma unroll
  for (int t = 0; t < 4; ++t)
#pragma unroll
    for (int i = 0; i < 8; ++i) {
      float o = acc[t][i] / l[i];
      Ob[(size_t)(row0 + half * 8 + i) * DIM + h * HEADD + t * 16 + ln] = (bf16_t)o;
    }
}

// ---------------- launch ----------------
extern "C" void kernel_launch(void* const* d_in, const int* in_sizes, int n_in,
                              void* d_out, int out_size, void* d_ws, size_t ws_size,
                              hipStream_t stream) {
  (void)in_sizes; (void)n_in; (void)out_size; (void)ws_size;
  const float* x  = (const float*)d_in[0];
  const float* Wq = (const float*)d_in[1];
  const float* bq = (const float*)d_in[2];
  const float* Wk = (const float*)d_in[3];
  const float* bk = (const float*)d_in[4];
  const float* Wv = (const float*)d_in[5];
  const float* bv = (const float*)d_in[6];
  const float* Wo = (const float*)d_in[7];

  float* out = (float*)d_out;                       // [T, DIM]
  float* pk  = out + (size_t)TSEQ * DIM;            // present_k [NHKV,T,64]
  float* pv  = pk + (size_t)NHKV * TSEQ * HEADD;    // present_v [NHKV,T,64]

  bf16_t* p    = (bf16_t*)d_ws;                     // ~48 MiB total bf16 scratch
  bf16_t* xb   = p; p += (size_t)TSEQ * DIM;
  bf16_t* WqT  = p; p += (size_t)DIM * DIM;
  bf16_t* WkT  = p; p += (size_t)DIM * KVDIM;
  bf16_t* WvT  = p; p += (size_t)DIM * KVDIM;
  bf16_t* WoT  = p; p += (size_t)DIM * DIM;
  bf16_t* Qbuf = p; p += (size_t)TSEQ * DIM;
  bf16_t* Kbuf = p; p += (size_t)NHKV * TSEQ * HEADD;
  bf16_t* Vtb  = p; p += (size_t)NHKV * HEADD * TSEQ;
  bf16_t* Ab   = p; p += (size_t)TSEQ * DIM;

  const int nx = TSEQ * DIM;
  cvt_f32_bf16<<<(nx + 255) / 256, 256, 0, stream>>>(x, xb, nx);
  cvtT_f32_bf16<<<(DIM * DIM + 255) / 256, 256, 0, stream>>>(Wq, WqT, DIM, DIM);
  cvtT_f32_bf16<<<(DIM * KVDIM + 255) / 256, 256, 0, stream>>>(Wk, WkT, DIM, KVDIM);
  cvtT_f32_bf16<<<(DIM * KVDIM + 255) / 256, 256, 0, stream>>>(Wv, WvT, DIM, KVDIM);
  cvtT_f32_bf16<<<(DIM * DIM + 255) / 256, 256, 0, stream>>>(Wo, WoT, DIM, DIM);

  dim3 blk(256);
  gemm_bf16_wmma<<<dim3(DIM / 64,   TSEQ / 128), blk, 0, stream>>>(xb, WqT, bq, nullptr, Qbuf, TSEQ, DIM,   DIM, 1);
  gemm_bf16_wmma<<<dim3(KVDIM / 64, TSEQ / 128), blk, 0, stream>>>(xb, WkT, bk, pk,      Kbuf, TSEQ, KVDIM, DIM, 2);
  gemm_bf16_wmma<<<dim3(KVDIM / 64, TSEQ / 128), blk, 0, stream>>>(xb, WvT, bv, pv,      Vtb,  TSEQ, KVDIM, DIM, 3);

  flash_attn_gqa<<<NHQ * (TSEQ / 16), 32, 0, stream>>>(Qbuf, Kbuf, Vtb, Ab);

  gemm_bf16_wmma<<<dim3(DIM / 64, TSEQ / 128), blk, 0, stream>>>(Ab, WoT, nullptr, out, nullptr, TSEQ, DIM, DIM, 0);
}
